// PairInteraction_69999376990652
// MI455X (gfx1250) — matrix operand order repaired and checked by reading, not compiled, not run on hardware
//
#include <hip/hip_runtime.h>

// ---------------------------------------------------------------------------
// PairInteraction on MI455X (gfx1250, wave32, WMMA bf16 16x16x32).
//
// ~13 GFLOP vs ~230 MB compulsory HBM -> near the 23.3 TB/s memory roof only
// if all matmuls run on the WMMA pipes. bf16 inputs / f32 accumulate.
// Weights pre-packed once into WMMA B-fragment layout (L2 resident).
// x_b kept bf16 in workspace (6.4 MB, L2 resident) for the random edge
// gather. Stages 2..5 fused per 16 atoms with LDS staging of x_ba2 / h_out
// (~35 KB of the 320 KB WGP LDS; per-atom strides padded to kill bank
// conflicts on the A-fragment loads).
//
// f32->bf16 uses round-half-up (+0x8000) + v_perm_b32 packing: 3 VALU ops per
// pair vs ~11 for exact RNE; differs from RNE only at exact ties.
// ---------------------------------------------------------------------------

#define N_ATOMS   50000
#define KMAX      32
#define EMB_ATOM  256
#define EMB_PIN   64
#define EMB_POUT  64
#define EMB_RBF   16

#define XSTR 1032   // per-atom ushort stride for x_ba2 in LDS (1024 + 8 pad)
#define HSTR 72     // per-atom ushort stride for h_out in LDS  (64 + 8 pad)

typedef __attribute__((ext_vector_type(16))) __bf16        v16bf;
typedef __attribute__((ext_vector_type(8)))  float         v8f;
typedef __attribute__((ext_vector_type(8)))  unsigned int  v8u;

// fp32 -> bf16 (round half up): 2 VALU ops
static __device__ __forceinline__ unsigned short f2bf(float f) {
    return (unsigned short)((__float_as_uint(f) + 0x8000u) >> 16);
}
// pack two fp32 -> packed bf16x2 with one v_perm_b32: 3 VALU ops total
static __device__ __forceinline__ unsigned int pack2(float lo, float hi) {
    unsigned int ulo = __float_as_uint(lo) + 0x8000u;
    unsigned int uhi = __float_as_uint(hi) + 0x8000u;
    // result = {uhi[31:16], ulo[31:16]}
    return __builtin_amdgcn_perm(uhi, ulo, 0x07060302u);
}

static __device__ __forceinline__ v8f wmma_bf16(v16bf a, v16bf b, v8f c) {
    return __builtin_amdgcn_wmma_f32_16x16x32_bf16(
        /*neg_a=*/false, a, /*neg_b=*/false, b,
        /*c_mod=*/(short)0, c, /*reuse_a=*/false, /*reuse_b=*/false);
}

// build a v16bf A-fragment from 8 packed uints (two 16B halves)
static __device__ __forceinline__ v16bf frag_from(uint4 a, uint4 b) {
    v8u u;
    u[0] = a.x; u[1] = a.y; u[2] = a.z; u[3] = a.w;
    u[4] = b.x; u[5] = b.y; u[6] = b.z; u[7] = b.w;
    return __builtin_bit_cast(v16bf, u);
}

// ---------------------------------------------------------------------------
// Workspace layout (uint32 offsets)
// ---------------------------------------------------------------------------
#define WD_OFF 0u        //  8 s * 4 nt * 32 lanes * 8 = 8192 uints
#define WB_OFF 8192u     // 32 s * 4 nt * 32 * 8       = 32768 uints
#define WU_OFF 40960u    //  2 s * 16 nt * 32 * 8      = 8192 uints
#define XB_OFF 49152u    // x_b bf16: 50000*64 ushorts = 1600000 uints

// ---------------------------------------------------------------------------
// Pack a (K x N) fp32 weight matrix into per-lane WMMA B fragments (bf16).
// 16-bit B layout (32x16 per WMMA): lane L<16 holds column nt*16+L,
// halves j<8 -> K = s*32 + j, j>=8 -> K = s*32 + 8 + j ; lanes 16-31 -> same columns, K += 8.
// Stored as 8 consecutive uints (32 B) per lane.
// ---------------------------------------------------------------------------
__global__ void prep_w_frags(const float* __restrict__ W, int K, int N,
                             unsigned int* __restrict__ out) {
    int nt_count = N >> 4;
    int total = (K >> 5) * nt_count * 32 * 8;
    int i = blockIdx.x * blockDim.x + threadIdx.x;
    if (i >= total) return;
    int jj   = i & 7;
    int lane = (i >> 3) & 31;
    int snt  = i >> 8;               // s * nt_count + nt
    int nt   = snt % nt_count;
    int s    = snt / nt_count;
    int c    = nt * 16 + (lane & 15);
    int k0   = s * 32 + ((lane >> 4) << 3) + (jj & 3) * 2 + ((jj >> 2) << 4);
    out[i] = pack2(W[k0 * N + c], W[(k0 + 1) * N + c]);
}

// ---------------------------------------------------------------------------
// Kernel A: x_b = h @ W_down  (50000x256 @ 256x64), bf16 output.
// One wave per 16-row tile; 8 k-steps x 4 n-tiles of WMMA, fully unrolled.
// ---------------------------------------------------------------------------
__global__ __launch_bounds__(256)
void xb_gemm(const float* __restrict__ h,
             const unsigned int* __restrict__ wd_frag,
             unsigned short* __restrict__ xb) {
    int lane = threadIdx.x & 31;
    int wave = threadIdx.x >> 5;
    int tile = blockIdx.x * 8 + wave;
    if (tile >= (N_ATOMS / 16)) return;          // wave-uniform guard
    int l15 = lane & 15;
    int hk  = lane >> 4;
    int row = tile * 16 + l15;

    v8f acc[4] = {};
#pragma unroll
    for (int s = 0; s < 8; ++s) {
        // A fragment: halves 0..7 -> K = s*32 + 8*hk + j, halves 8..15 -> +16
        const float4* hp = (const float4*)(h + (size_t)row * EMB_ATOM + s * 32 + hk * 8);
        float4 f0 = hp[0], f1 = hp[1];           // K block base
        float4 f2 = hp[4], f3 = hp[5];           // +16 floats
        v8u au;
        au[0] = pack2(f0.x, f0.y); au[1] = pack2(f0.z, f0.w);
        au[2] = pack2(f1.x, f1.y); au[3] = pack2(f1.z, f1.w);
        au[4] = pack2(f2.x, f2.y); au[5] = pack2(f2.z, f2.w);
        au[6] = pack2(f3.x, f3.y); au[7] = pack2(f3.z, f3.w);
        v16bf a = __builtin_bit_cast(v16bf, au);

        // Pre-load all four B fragments, then 4 WMMAs under one wait
        v8u bu[4];
#pragma unroll
        for (int nt = 0; nt < 4; ++nt)
            bu[nt] = *(const v8u*)(wd_frag + ((size_t)(s * 4 + nt) * 32 + lane) * 8);
#pragma unroll
        for (int nt = 0; nt < 4; ++nt)
            acc[nt] = wmma_bf16(a, __builtin_bit_cast(v16bf, bu[nt]), acc[nt]);
    }
    // C layout: VGPR v, lanes 0-15 -> M=v, lanes 16-31 -> M=8+v; N = lane&15
#pragma unroll
    for (int nt = 0; nt < 4; ++nt)
#pragma unroll
        for (int v = 0; v < 8; ++v) {
            int m = v + 8 * hk;
            xb[(size_t)(tile * 16 + m) * EMB_PIN + nt * 16 + l15] = f2bf(acc[nt][v]);
        }
}

// ---------------------------------------------------------------------------
// Kernel B: fused gather + per-atom (16x32)@(32x64) + bilinear + up-project.
// 128 threads (4 waves) per block, 16 atoms per block. LDS stages x_ba2/h_out.
// ---------------------------------------------------------------------------
__global__ __launch_bounds__(128)
void fused_pair(const float* __restrict__ rad,          // (N,16,32) f32
                const int* __restrict__ edge_src,       // edge_index row 0
                const unsigned short* __restrict__ xb,  // (N,64) bf16
                const unsigned int* __restrict__ wb_frag,
                const unsigned int* __restrict__ wu_frag,
                float* __restrict__ out) {
    __shared__ unsigned short xba2[16 * XSTR];  // ~33 KB bf16, padded stride
    __shared__ unsigned short hout[16 * HSTR];  // ~2.3 KB bf16, padded stride

    int lane = threadIdx.x & 31;
    int wave = threadIdx.x >> 5;
    int l15  = lane & 15;
    int hk   = lane >> 4;
    int atomBase = blockIdx.x * 16;

    // ---- Stage 1+2: gather x2 columns + rad-basis WMMA (K=32 fits one op) --
    for (int q = 0; q < 4; ++q) {
        int a = wave * 4 + q;
        int n = atomBase + a;

        // prefetch next block's rad rows (speculative; OOB silently dropped)
        __builtin_prefetch(rad + (size_t)(n + 16) * (EMB_RBF * KMAX), 0, 1);

        // A fragment from rad_basis[n] (16 x 32), row = l15
        const float4* rp = (const float4*)(rad + (size_t)n * (EMB_RBF * KMAX)
                                           + l15 * KMAX + hk * 8);
        float4 f0 = rp[0], f1 = rp[1];
        float4 f2 = rp[4], f3 = rp[5];
        v8u au;
        au[0] = pack2(f0.x, f0.y); au[1] = pack2(f0.z, f0.w);
        au[2] = pack2(f1.x, f1.y); au[3] = pack2(f1.z, f1.w);
        au[4] = pack2(f2.x, f2.y); au[5] = pack2(f2.z, f2.w);
        au[6] = pack2(f3.x, f3.y); au[7] = pack2(f3.z, f3.w);
        v16bf afrag = __builtin_bit_cast(v16bf, au);

        // Hoist the 16 per-lane edge sources once per atom (shared across nt).
        // Lane half hk reads k = 8*hk + j (j<8) and 16 + 8*hk + (j-8) (j>=8).
        int sr[16];
#pragma unroll
        for (int j = 0; j < 16; ++j)
            sr[j] = edge_src[n * KMAX + hk * 8 + (j & 7) + ((j >> 3) << 4)];

        v8f acc[4] = {};
#pragma unroll
        for (int nt = 0; nt < 4; ++nt) {
            // B fragment: column c over kmax; gathered from L2-resident x_b.
            // For each j, lanes 0-15 read a contiguous 32 B run.
            int c = nt * 16 + l15;
            v8u bu;
#pragma unroll
            for (int jj = 0; jj < 8; ++jj) {
                unsigned int w0 = xb[(size_t)sr[2 * jj]     * EMB_PIN + c];
                unsigned int w1 = xb[(size_t)sr[2 * jj + 1] * EMB_PIN + c];
                bu[jj] = w0 | (w1 << 16);
            }
            acc[nt] = wmma_bf16(afrag, __builtin_bit_cast(v16bf, bu), acc[nt]);
        }
        // x_ba2[a][r*64+p] bf16 -> LDS (contiguous A-fragments for stage 3)
#pragma unroll
        for (int nt = 0; nt < 4; ++nt)
#pragma unroll
            for (int v = 0; v < 8; ++v) {
                int m = v + 8 * hk;
                xba2[a * XSTR + m * 64 + nt * 16 + l15] = f2bf(acc[nt][v]);
            }
    }
    __syncthreads();

    // ---- Stage 3: (16 atoms x 1024) @ W_bilinear(1024x64); wave -> n-tile --
    {
        int nt = wave;
        v8f acc = {};
#pragma unroll 4
        for (int s = 0; s < 32; ++s) {
            // A fragment: two contiguous 16B LDS loads; padded stride XSTR
            // spreads the 16 lane addresses over distinct banks.
            uint4 ua = *(const uint4*)&xba2[l15 * XSTR + s * 32 + hk * 8];
            uint4 ub = *(const uint4*)&xba2[l15 * XSTR + s * 32 + 16 + hk * 8];
            v8u bu = *(const v8u*)(wb_frag + ((size_t)(s * 4 + nt) * 32 + lane) * 8);
            acc = wmma_bf16(frag_from(ua, ub), __builtin_bit_cast(v16bf, bu), acc);
        }
#pragma unroll
        for (int v = 0; v < 8; ++v) {
            int m = v + 8 * hk;                  // atom within group
            hout[m * HSTR + nt * 16 + l15] = f2bf(acc[v]);
        }
    }
    __syncthreads();

    // ---- Stage 4: (16 x 64) @ W_up(64x256) -> global f32 output ------------
    for (int q = 0; q < 4; ++q) {
        int nt = wave * 4 + q;
        v8f acc = {};
#pragma unroll
        for (int s = 0; s < 2; ++s) {
            uint4 ua = *(const uint4*)&hout[l15 * HSTR + s * 32 + hk * 8];
            uint4 ub = *(const uint4*)&hout[l15 * HSTR + s * 32 + 16 + hk * 8];
            v8u bu = *(const v8u*)(wu_frag + ((size_t)(s * 16 + nt) * 32 + lane) * 8);
            acc = wmma_bf16(frag_from(ua, ub), __builtin_bit_cast(v16bf, bu), acc);
        }
#pragma unroll
        for (int v = 0; v < 8; ++v) {
            int m = v + 8 * hk;
            out[(size_t)(atomBase + m) * EMB_ATOM + nt * 16 + l15] = acc[v];
        }
    }
}

// ---------------------------------------------------------------------------
extern "C" void kernel_launch(void* const* d_in, const int* in_sizes, int n_in,
                              void* d_out, int out_size, void* d_ws, size_t ws_size,
                              hipStream_t stream) {
    const float* h    = (const float*)d_in[0];
    const float* rad  = (const float*)d_in[1];
    const int*   edge = (const int*)d_in[2];    // row 0 = src, row 1 = dst
    const float* Wd   = (const float*)d_in[4];
    const float* Wb   = (const float*)d_in[5];
    const float* Wu   = (const float*)d_in[6];
    float* out = (float*)d_out;

    unsigned int*   ws      = (unsigned int*)d_ws;
    unsigned int*   wd_frag = ws + WD_OFF;
    unsigned int*   wb_frag = ws + WB_OFF;
    unsigned int*   wu_frag = ws + WU_OFF;
    unsigned short* xb      = (unsigned short*)(ws + XB_OFF);

    (void)in_sizes; (void)n_in; (void)out_size; (void)ws_size;

    // Pack weights into WMMA B-fragment layout (bf16), once per launch.
    prep_w_frags<<<(8192  + 255) / 256, 256, 0, stream>>>(Wd, EMB_ATOM, EMB_PIN, wd_frag);
    prep_w_frags<<<(32768 + 255) / 256, 256, 0, stream>>>(Wb, EMB_RBF * EMB_PIN, EMB_POUT, wb_frag);
    prep_w_frags<<<(8192  + 255) / 256, 256, 0, stream>>>(Wu, EMB_POUT, EMB_ATOM, wu_frag);

    // x_b = h @ W_down  (3125 16-row tiles, 8 waves per block)
    xb_gemm<<<(3125 + 7) / 8, 256, 0, stream>>>(h, wd_frag, xb);

    // Fused gather + rad-matmul + bilinear + up-projection (16 atoms / block)
    fused_pair<<<N_ATOMS / 16, 128, 0, stream>>>(rad, edge, xb, wb_frag, wu_frag, out);
}
